// CandidateRelationalActionQRanker_51823075393570
// MI455X (gfx1250) — matrix operand order repaired and checked by reading, not compiled, not run on hardware
//
#include <hip/hip_runtime.h>
#include <hip/hip_bf16.h>
#include <math.h>

// ---------------------------------------------------------------------------
// CDNA5 (gfx1250) types for WMMA
// ---------------------------------------------------------------------------
typedef __attribute__((ext_vector_type(16))) __bf16 v16bf;
typedef __attribute__((ext_vector_type(8)))  float  v8f;

#define N_NODES 1024
#define HID     64

// A-fragment K mapping for 16-bit A 16x32 (contiguous, matches B):
// lane<16: K=e ;  lane>=16: K=e+16
__device__ __forceinline__ int a_kmap(int e, int hi) {
    return e + 16 * hi;
}

// |x| via an opaque VALU op. Using fabsf() here triggers an LLVM combine that
// folds the sign-mask across the packed bf16 conversion and crashes gfx1250
// ISel (malformed i32 = and(bitcast v2f32)). Inline asm is a combine barrier.
__device__ __forceinline__ float abs_opaque(float x) {
    float r;
    asm("v_and_b32 %0, 0x7fffffff, %1" : "=v"(r) : "v"(x));
    return r;
}

// ---------------------------------------------------------------------------
// Generic small helpers (cheap stages: < 0.5 GFLOP total, VALU is fine)
// ---------------------------------------------------------------------------
__global__ void zero_kernel(float* p, int n) {
    int i = blockIdx.x * blockDim.x + threadIdx.x;
    if (i < n) p[i] = 0.f;
}

// C[M,N] = A[M,K] @ W[K,N] + b   (row-major), optional relu
__global__ void gemm_bias_kernel(const float* __restrict__ A,
                                 const float* __restrict__ W,
                                 const float* __restrict__ b,
                                 float* __restrict__ C,
                                 int M, int N, int K, int relu) {
    int idx = blockIdx.x * blockDim.x + threadIdx.x;
    if (idx >= M * N) return;
    int m = idx / N, n = idx % N;
    float acc = b ? b[n] : 0.f;
    const float* a = A + m * K;
    for (int k = 0; k < K; ++k) acc += a[k] * W[k * N + n];
    if (relu) acc = fmaxf(acc, 0.f);
    C[idx] = acc;
}

// y[row] = LN(x[row] (+ res[row])) * g + b, optional relu.  64-wide rows.
__global__ void ln64_kernel(const float* __restrict__ x,
                            const float* __restrict__ res,
                            const float* __restrict__ g,
                            const float* __restrict__ b,
                            float* __restrict__ y, int relu) {
    int row = blockIdx.x;
    int t = threadIdx.x;                 // 64 threads
    float v = x[row * HID + t] + (res ? res[row * HID + t] : 0.f);
    __shared__ float sb[HID];
    sb[t] = v;
    __syncthreads();
    for (int s = 32; s > 0; s >>= 1) { if (t < s) sb[t] += sb[t + s]; __syncthreads(); }
    float mean = sb[0] * (1.f / HID);
    __syncthreads();
    float d = v - mean;
    sb[t] = d * d;
    __syncthreads();
    for (int s = 32; s > 0; s >>= 1) { if (t < s) sb[t] += sb[t + s]; __syncthreads(); }
    float var = sb[0] * (1.f / HID);
    float o = d * rsqrtf(var + 1e-5f) * g[t] + b[t];
    if (relu) o = fmaxf(o, 0.f);
    y[row * HID + t] = o;
}

// 4-head attention, HD=16, seq=1024.  One block per (i, head).
__global__ void attn_kernel(const float* __restrict__ qkv, float* __restrict__ o) {
    int i = blockIdx.x, h = blockIdx.y;
    const float* q = qkv + i * 192 + h * 16;
    __shared__ float s[N_NODES];
    __shared__ float red[128];
    int t = threadIdx.x;                 // 128 threads
    for (int j = t; j < N_NODES; j += 128) {
        const float* kk = qkv + j * 192 + 64 + h * 16;
        float d = 0.f;
        #pragma unroll
        for (int e = 0; e < 16; ++e) d += q[e] * kk[e];
        s[j] = d * 0.25f;                // 1/sqrt(16)
    }
    __syncthreads();
    float m = -3.4e38f;
    for (int j = t; j < N_NODES; j += 128) m = fmaxf(m, s[j]);
    red[t] = m; __syncthreads();
    for (int st = 64; st > 0; st >>= 1) { if (t < st) red[t] = fmaxf(red[t], red[t + st]); __syncthreads(); }
    m = red[0]; __syncthreads();
    float sum = 0.f;
    for (int j = t; j < N_NODES; j += 128) { float e = __expf(s[j] - m); s[j] = e; sum += e; }
    red[t] = sum; __syncthreads();
    for (int st = 64; st > 0; st >>= 1) { if (t < st) red[t] += red[t + st]; __syncthreads(); }
    float inv = 1.f / red[0];
    __syncthreads();
    int d = t & 15, grp = t >> 4;        // 8 groups of 16
    float acc = 0.f;
    for (int j = grp; j < N_NODES; j += 8) acc += s[j] * qkv[j * 192 + 128 + h * 16 + d];
    red[t] = acc; __syncthreads();
    if (grp == 0) {
        float a = 0.f;
        #pragma unroll
        for (int gg = 0; gg < 8; ++gg) a += red[gg * 16 + d];
        o[i * HID + h * 16 + d] = a * inv;
    }
}

// Fold lhs/rhs/diff terms of the pair MLP into per-node precomputes:
//   A[i] = enc_i @ (W_lhs + W_diff) + b1 ;  B[j] = enc_j @ (W_rhs - W_diff)
__global__ void prep_pair_kernel(const float* __restrict__ enc,
                                 const float* __restrict__ w1,   // [256,64]
                                 const float* __restrict__ b1,
                                 float* __restrict__ A, float* __restrict__ B) {
    int idx = blockIdx.x * blockDim.x + threadIdx.x;
    if (idx >= N_NODES * HID) return;
    int i = idx >> 6, c = idx & 63;
    float a = b1[c], b = 0.f;
    for (int k = 0; k < HID; ++k) {
        float e = enc[i * HID + k];
        float wd = w1[(128 + k) * HID + c];
        a += e * (w1[k * HID + c] + wd);
        b += e * (w1[(64 + k) * HID + c] - wd);
    }
    A[idx] = a;
    B[idx] = b;
}

// Row-wise softmax in place over 1024-wide rows (gates, 4 MB: L2-resident)
__global__ void softmax_row_kernel(float* __restrict__ g) {
    int row = blockIdx.x;
    float* r = g + row * N_NODES;
    __shared__ float sb[256];
    int t = threadIdx.x;
    float m = -3.4e38f;
    for (int j = t; j < N_NODES; j += 256) m = fmaxf(m, r[j]);
    sb[t] = m; __syncthreads();
    for (int s = 128; s > 0; s >>= 1) { if (t < s) sb[t] = fmaxf(sb[t], sb[t + s]); __syncthreads(); }
    m = sb[0]; __syncthreads();
    float sum = 0.f;
    for (int j = t; j < N_NODES; j += 256) { float e = __expf(r[j] - m); r[j] = e; sum += e; }
    sb[t] = sum; __syncthreads();
    for (int s = 128; s > 0; s >>= 1) { if (t < s) sb[t] += sb[t + s]; __syncthreads(); }
    float inv = 1.f / sb[0];
    for (int j = t; j < N_NODES; j += 256) r[j] *= inv;
}

// ---------------------------------------------------------------------------
// Pair stage kernel 1: gates[i,j].
// One wave per i; 16 j's per WMMA tile. X = |enc_i - enc_j| built straight
// into A-fragment layout; 8x v_wmma_f32_16x16x32_bf16 per tile.
// ---------------------------------------------------------------------------
__global__ __launch_bounds__(256) void pair_gates_kernel(
    const float* __restrict__ enc,
    const float* __restrict__ pg_w1,   // [256,64]
    const float* __restrict__ pg_w2,   // [64]
    const float* __restrict__ pg_b2,   // [1]
    const float* __restrict__ Ag,      // [1024,64]
    const float* __restrict__ Bg,      // [1024,64]
    float* __restrict__ gates)         // [1024,1024]
{
    const int lane = threadIdx.x & 31;
    const int wid  = threadIdx.x >> 5;
    const int i    = blockIdx.x * 8 + wid;
    const int hi   = lane >> 4;
    const int col  = lane & 15;

    // B-fragments of W_abs (rows 192..255 of pg_w1), kept in registers
    const float* Wabs = pg_w1 + 192 * HID;
    v16bf bw[2][4];
    #pragma unroll
    for (int c = 0; c < 2; ++c)
        #pragma unroll
        for (int t = 0; t < 4; ++t) {
            v16bf b;
            #pragma unroll
            for (int e = 0; e < 16; ++e) {
                int k = 32 * c + 16 * hi + e;          // B layout: K striped across halves
                b[e] = (__bf16)Wabs[k * HID + (col + 16 * t)];
            }
            bw[c][t] = b;
        }

    // enc_i sampled at this lane's A-fragment K positions (reused every tile)
    float ei[32];
    #pragma unroll
    for (int e = 0; e < 16; ++e) {
        int kl = a_kmap(e, hi);
        ei[e]      = enc[i * HID + kl];
        ei[16 + e] = enc[i * HID + 32 + kl];
    }
    float agv[4], w2l[4];
    #pragma unroll
    for (int t = 0; t < 4; ++t) {
        agv[t] = Ag[i * HID + col + 16 * t];
        w2l[t] = pg_w2[col + 16 * t];
    }
    const float b2 = pg_b2[0];

    const int jt0 = blockIdx.y * 16;
    for (int jt = jt0; jt < jt0 + 16; ++jt) {
        const int j0 = jt * 16;
        const int jr = j0 + col;                       // A row this lane feeds
        v16bf a0, a1;
        #pragma unroll
        for (int e = 0; e < 16; ++e) {
            int kl = a_kmap(e, hi);
            a0[e] = (__bf16)abs_opaque(enc[jr * HID + kl]      - ei[e]);
            a1[e] = (__bf16)abs_opaque(enc[jr * HID + 32 + kl] - ei[16 + e]);
        }
        v8f cc[4];
        #pragma unroll
        for (int t = 0; t < 4; ++t) {
            v8f c = {};
            c = __builtin_amdgcn_wmma_f32_16x16x32_bf16(false, a0, false, bw[0][t], (short)0, c, false, false);
            c = __builtin_amdgcn_wmma_f32_16x16x32_bf16(false, a1, false, bw[1][t], (short)0, c, false, false);
            cc[t] = c;
        }
        // gate_j = relu(hidden) . pg_w2 ; reduce across the 16 lanes of a half
        #pragma unroll
        for (int r = 0; r < 8; ++r) {
            int jrow = j0 + r + 8 * hi;                // C row = r + 8*half
            float p = 0.f;
            #pragma unroll
            for (int t = 0; t < 4; ++t) {
                float hv = cc[t][r] + agv[t] + Bg[jrow * HID + col + 16 * t];
                p += fmaxf(hv, 0.f) * w2l[t];
            }
            p += __shfl_xor(p, 8, 16);
            p += __shfl_xor(p, 4, 16);
            p += __shfl_xor(p, 2, 16);
            p += __shfl_xor(p, 1, 16);
            if (col == 0) {
                float gv = p + b2;
                if (jrow == i) gv = -1e9f;             // diagonal mask pre-softmax
                gates[i * N_NODES + jrow] = gv;
            }
        }
    }
}

// ---------------------------------------------------------------------------
// Pair stage kernel 2: pair_context[i] = sum_j w[i,j] * vals[i,j,:].
// Same X WMMAs, relu'd hidden bounced through LDS into A layout, second
// 64->64 WMMA layer, fused weighted accumulation (softmax weights sum to 1,
// so pv_b2 is added exactly once).
// ---------------------------------------------------------------------------
__global__ __launch_bounds__(256) void pair_vals_kernel(
    const float* __restrict__ enc,
    const float* __restrict__ pv_w1,   // [256,64]
    const float* __restrict__ pv_w2,   // [64,64]
    const float* __restrict__ pv_b2,   // [64]
    const float* __restrict__ Av,
    const float* __restrict__ Bv,
    const float* __restrict__ wsoft,   // softmaxed gates [1024,1024]
    float* __restrict__ pctx)          // [1024,64], atomically accumulated
{
    const int lane = threadIdx.x & 31;
    const int wid  = threadIdx.x >> 5;
    const int i    = blockIdx.x * 8 + wid;
    const int hi   = lane >> 4;
    const int col  = lane & 15;

    __shared__ __bf16 hbuf[8][16][72];                 // per-wave hidden tile
    __bf16 (*hb)[72] = hbuf[wid];

    const float* Wabs = pv_w1 + 192 * HID;
    v16bf bw[2][4], b2w[2][4];
    #pragma unroll
    for (int c = 0; c < 2; ++c)
        #pragma unroll
        for (int t = 0; t < 4; ++t) {
            v16bf b, b2f;
            #pragma unroll
            for (int e = 0; e < 16; ++e) {
                int k = 32 * c + 16 * hi + e;
                b[e]   = (__bf16)Wabs[k * HID + (col + 16 * t)];
                b2f[e] = (__bf16)pv_w2[k * HID + (col + 16 * t)];
            }
            bw[c][t] = b;
            b2w[c][t] = b2f;
        }

    float ei[32];
    #pragma unroll
    for (int e = 0; e < 16; ++e) {
        int kl = a_kmap(e, hi);
        ei[e]      = enc[i * HID + kl];
        ei[16 + e] = enc[i * HID + 32 + kl];
    }
    float avv[4], b2c[4];
    #pragma unroll
    for (int t = 0; t < 4; ++t) {
        avv[t] = Av[i * HID + col + 16 * t];
        b2c[t] = pv_b2[col + 16 * t];
    }

    float acc[4] = {0.f, 0.f, 0.f, 0.f};

    const int jt0 = blockIdx.y * 16;
    for (int jt = jt0; jt < jt0 + 16; ++jt) {
        const int j0 = jt * 16;
        const int jr = j0 + col;
        v16bf a0, a1;
        #pragma unroll
        for (int e = 0; e < 16; ++e) {
            int kl = a_kmap(e, hi);
            a0[e] = (__bf16)abs_opaque(enc[jr * HID + kl]      - ei[e]);
            a1[e] = (__bf16)abs_opaque(enc[jr * HID + 32 + kl] - ei[16 + e]);
        }
        // layer 1: hidden_v = relu(|diff| @ Wabs + Av_i + Bv_j)  -> LDS (bf16)
        #pragma unroll
        for (int t = 0; t < 4; ++t) {
            v8f c = {};
            c = __builtin_amdgcn_wmma_f32_16x16x32_bf16(false, a0, false, bw[0][t], (short)0, c, false, false);
            c = __builtin_amdgcn_wmma_f32_16x16x32_bf16(false, a1, false, bw[1][t], (short)0, c, false, false);
            #pragma unroll
            for (int r = 0; r < 8; ++r) {
                int jrow = j0 + r + 8 * hi;
                float hv = c[r] + avv[t] + Bv[jrow * HID + col + 16 * t];
                hb[r + 8 * hi][col + 16 * t] = (__bf16)fmaxf(hv, 0.f);
            }
        }
        __syncthreads();                               // LDS write -> cross-lane read
        // layer 2: vals = hidden_v @ pv_w2  (re-loaded in A-fragment layout)
        v16bf h0, h1;
        #pragma unroll
        for (int e = 0; e < 16; ++e) {
            int kl = a_kmap(e, hi);
            h0[e] = hb[col][kl];
            h1[e] = hb[col][32 + kl];
        }
        #pragma unroll
        for (int t = 0; t < 4; ++t) {
            v8f v = {};
            v = __builtin_amdgcn_wmma_f32_16x16x32_bf16(false, h0, false, b2w[0][t], (short)0, v, false, false);
            v = __builtin_amdgcn_wmma_f32_16x16x32_bf16(false, h1, false, b2w[1][t], (short)0, v, false, false);
            #pragma unroll
            for (int r = 0; r < 8; ++r) {
                float w = wsoft[i * N_NODES + j0 + r + 8 * hi];
                acc[t] += w * v[r];
            }
        }
        __syncthreads();                               // protect hbuf for next tile
    }

    // combine halves (rows 0-7 vs 8-15) and commit
    #pragma unroll
    for (int t = 0; t < 4; ++t) {
        float tot = acc[t] + __shfl_xor(acc[t], 16, 32);
        if (hi == 0) {
            if (blockIdx.y == 0) tot += b2c[t];        // sum_j w == 1 -> bias once
            atomicAdd(&pctx[i * HID + col + 16 * t], tot);
        }
    }
}

// column mean over 1024 rows
__global__ void colmean_kernel(const float* __restrict__ x, float* __restrict__ out) {
    int c = blockIdx.x;
    __shared__ float sb[256];
    int t = threadIdx.x;
    float s = 0.f;
    for (int i = t; i < N_NODES; i += 256) s += x[i * HID + c];
    sb[t] = s; __syncthreads();
    for (int st = 128; st > 0; st >>= 1) { if (t < st) sb[t] += sb[t + st]; __syncthreads(); }
    if (t == 0) out[c] = sb[0] * (1.f / N_NODES);
}

// score head: hidden = relu([enc2, gmean] @ sh_w1 + sh_b1)
__global__ void head1_kernel(const float* __restrict__ enc2,
                             const float* __restrict__ gmean,
                             const float* __restrict__ w,   // [128,64]
                             const float* __restrict__ b,
                             float* __restrict__ hdn) {
    int idx = blockIdx.x * blockDim.x + threadIdx.x;
    if (idx >= N_NODES * HID) return;
    int i = idx >> 6, c = idx & 63;
    float acc = b[c];
    for (int k = 0; k < HID; ++k) acc += enc2[i * HID + k] * w[k * HID + c];
    for (int k = 0; k < HID; ++k) acc += gmean[k] * w[(HID + k) * HID + c];
    hdn[idx] = fmaxf(acc, 0.f);
}

__global__ void head2_kernel(const float* __restrict__ hdn,
                             const float* __restrict__ w2,  // [64]
                             const float* __restrict__ b2,  // [1]
                             float* __restrict__ out) {
    int i = blockIdx.x * blockDim.x + threadIdx.x;
    if (i >= N_NODES) return;
    float a = b2[0];
    for (int k = 0; k < HID; ++k) a += hdn[i * HID + k] * w2[k];
    out[i] = a;
}

// ---------------------------------------------------------------------------
extern "C" void kernel_launch(void* const* d_in, const int* in_sizes, int n_in,
                              void* d_out, int out_size, void* d_ws, size_t ws_size,
                              hipStream_t stream) {
    (void)in_sizes; (void)n_in; (void)out_size; (void)ws_size;
    const float* features   = (const float*)d_in[0];
    const float* ip_w1      = (const float*)d_in[1];
    const float* ip_b1      = (const float*)d_in[2];
    const float* ip_ln_g    = (const float*)d_in[3];
    const float* ip_ln_b    = (const float*)d_in[4];
    const float* ip_w2      = (const float*)d_in[5];
    const float* ip_b2      = (const float*)d_in[6];
    const float* attn_in_w  = (const float*)d_in[7];
    const float* attn_in_b  = (const float*)d_in[8];
    const float* attn_out_w = (const float*)d_in[9];
    const float* attn_out_b = (const float*)d_in[10];
    const float* attn_ln_g  = (const float*)d_in[11];
    const float* attn_ln_b  = (const float*)d_in[12];
    const float* pg_w1      = (const float*)d_in[13];
    const float* pg_b1      = (const float*)d_in[14];
    const float* pg_w2      = (const float*)d_in[15];
    const float* pg_b2      = (const float*)d_in[16];
    const float* pv_w1      = (const float*)d_in[17];
    const float* pv_b1      = (const float*)d_in[18];
    const float* pv_w2      = (const float*)d_in[19];
    const float* pv_b2      = (const float*)d_in[20];
    const float* ctx_ln_g   = (const float*)d_in[21];
    const float* ctx_ln_b   = (const float*)d_in[22];
    const float* sh_w1      = (const float*)d_in[23];
    const float* sh_b1      = (const float*)d_in[24];
    const float* sh_w2      = (const float*)d_in[25];
    const float* sh_b2      = (const float*)d_in[26];
    float* out = (float*)d_out;

    float* ws = (float*)d_ws;
    float* t0    = ws;                  // [1024,64]
    float* t1    = ws + 65536;          // [1024,64]
    float* h     = ws + 131072;         // [1024,64]
    float* qkv   = ws + 196608;         // [1024,192]
    float* o     = ws + 393216;         // [1024,64]
    float* t2    = ws + 458752;         // [1024,64]
    float* enc   = ws + 524288;         // [1024,64]
    float* Ag    = ws + 589824;         // [1024,64]
    float* Bg    = ws + 655360;
    float* Av    = ws + 720896;
    float* Bv    = ws + 786432;
    float* gates = ws + 851968;         // [1024,1024]
    float* pctx  = ws + 1900544;        // [1024,64]
    float* enc2  = ws + 1966080;        // [1024,64]
    float* gmean = ws + 2031616;        // [64]
    float* hdn   = ws + 2031680;        // [1024,64]

    const int NT = N_NODES * HID;       // 65536

    // input_proj: Linear -> LN -> ReLU -> Linear -> ReLU
    gemm_bias_kernel<<<NT / 256, 256, 0, stream>>>(features, ip_w1, ip_b1, t0, N_NODES, HID, 256, 0);
    ln64_kernel<<<N_NODES, HID, 0, stream>>>(t0, nullptr, ip_ln_g, ip_ln_b, t1, 1);
    gemm_bias_kernel<<<NT / 256, 256, 0, stream>>>(t1, ip_w2, ip_b2, h, N_NODES, HID, HID, 1);

    // attention + residual LN
    gemm_bias_kernel<<<(N_NODES * 192) / 256, 256, 0, stream>>>(h, attn_in_w, attn_in_b, qkv, N_NODES, 192, HID, 0);
    attn_kernel<<<dim3(N_NODES, 4), 128, 0, stream>>>(qkv, o);
    gemm_bias_kernel<<<NT / 256, 256, 0, stream>>>(o, attn_out_w, attn_out_b, t2, N_NODES, HID, HID, 0);
    ln64_kernel<<<N_NODES, HID, 0, stream>>>(t2, h, attn_ln_g, attn_ln_b, enc, 0);

    // fold lhs/rhs/diff into per-node precomputes
    prep_pair_kernel<<<NT / 256, 256, 0, stream>>>(enc, pg_w1, pg_b1, Ag, Bg);
    prep_pair_kernel<<<NT / 256, 256, 0, stream>>>(enc, pv_w1, pv_b1, Av, Bv);

    // all-pairs gates (WMMA) -> softmax -> values + weighted context (WMMA)
    pair_gates_kernel<<<dim3(128, 4), 256, 0, stream>>>(enc, pg_w1, pg_w2, pg_b2, Ag, Bg, gates);
    softmax_row_kernel<<<N_NODES, 256, 0, stream>>>(gates);
    zero_kernel<<<NT / 256, 256, 0, stream>>>(pctx, NT);
    pair_vals_kernel<<<dim3(128, 4), 256, 0, stream>>>(enc, pv_w1, pv_w2, pv_b2, Av, Bv, gates, pctx);

    // enc2 = LN(enc + pair_context); score head with mean context
    ln64_kernel<<<N_NODES, HID, 0, stream>>>(pctx, enc, ctx_ln_g, ctx_ln_b, enc2, 0);
    colmean_kernel<<<HID, 256, 0, stream>>>(enc2, gmean);
    head1_kernel<<<NT / 256, 256, 0, stream>>>(enc2, gmean, sh_w1, sh_b1, hdn);
    head2_kernel<<<N_NODES / 256, 256, 0, stream>>>(hdn, sh_w2, sh_b2, out);
}
